// CausalWanTransformer3DModel_78211354460598
// MI455X (gfx1250) — matrix operand order repaired and checked by reading, not compiled, not run on hardware
//
#include <hip/hip_runtime.h>

// ---------------------------------------------------------------------------
// CausalWan attention block on gfx1250 (CDNA5): bf16 WMMA + pipelined TDM.
// ---------------------------------------------------------------------------

typedef unsigned short u16;
typedef __attribute__((ext_vector_type(16))) __bf16        v16bf;
typedef __attribute__((ext_vector_type(8)))  float         v8f;
typedef __attribute__((ext_vector_type(4)))  unsigned int  u32x4;
typedef __attribute__((ext_vector_type(8)))  int           i32x8;
typedef __attribute__((ext_vector_type(4)))  int           i32x4;

union FragU { v16bf v; u32x4 q[2]; };

#define SEQ_   3120
#define DIM_   1536
#define HEADS_ 12
#define HD_    128
#define FRAME_ 1560
#define NQKV_  4608

#if defined(__HIP_DEVICE_COMPILE__) && __has_builtin(__builtin_amdgcn_tensor_load_to_lds)
#define USE_TDM 1
#else
#define USE_TDM 0
#endif

__device__ __forceinline__ int imin(int a, int b) { return a < b ? a : b; }

__device__ __forceinline__ u16 f2bf(float f) {
  union { float f; unsigned int u; } c; c.f = f;
  unsigned int u = c.u;
  unsigned int r = u + 0x7FFFu + ((u >> 16) & 1u);
  return (u16)(r >> 16);
}

#if USE_TDM
// Program a 2D Tensor-DMA descriptor (D#) and issue TENSOR_LOAD_TO_LDS.
// dim0 = contiguous elements (bf16), dim1 = rows; stride0 in elements.
// OOB rows (>= tensor_d1) read as zero -> free edge padding.
__device__ __forceinline__ void tdm_load_2d(unsigned lds_off, const void* gaddr,
                                            unsigned tile_d0, unsigned tile_d1,
                                            unsigned tensor_d0, unsigned tensor_d1,
                                            unsigned long long stride0) {
  unsigned long long ga = (unsigned long long)(uintptr_t)gaddr;
  u32x4 g0;
  g0[0] = 1u;                                          // count=1 (user D#)
  g0[1] = lds_off;                                     // lds_addr (bytes)
  g0[2] = (unsigned)(ga & 0xFFFFFFFFu);                // global_addr[31:0]
  g0[3] = (unsigned)((ga >> 32) & 0x01FFFFFFu) | (2u << 30);  // ga[56:32], type=2
  i32x8 g1;
  g1[0] = (int)(1u << 16);                             // data_size=1 -> 2 bytes
  g1[1] = (int)((tensor_d0 & 0xFFFFu) << 16);          // tensor_dim0[15:0]
  g1[2] = (int)(((tensor_d0 >> 16) & 0xFFFFu) | ((tensor_d1 & 0xFFFFu) << 16));
  g1[3] = (int)(((tensor_d1 >> 16) & 0xFFFFu) | ((tile_d0 & 0xFFFFu) << 16));
  g1[4] = (int)(tile_d1 & 0xFFFFu);                    // tile_dim1, tile_dim2=0
  g1[5] = (int)(unsigned)(stride0 & 0xFFFFFFFFull);    // dim0_stride[31:0]
  g1[6] = (int)(unsigned)((stride0 >> 32) & 0xFFFFull);// dim0_stride[47:32]
  g1[7] = 0;
  i32x4 z4 = {0, 0, 0, 0};
  i32x8 z8 = {0, 0, 0, 0, 0, 0, 0, 0};
  __builtin_amdgcn_tensor_load_to_lds(g0, g1, z4, z4, z8, 0);
}
__device__ __forceinline__ unsigned lds_off_u32(const void* p) {
  return (unsigned)(uintptr_t)p;                        // low 32 bits = LDS offset
}
#endif

// --------------------------- fp32 -> bf16 conversion -----------------------
__global__ void __launch_bounds__(256) cvt_bf16_kernel(const float* __restrict__ in,
                                                       u16* __restrict__ out, int n) {
  int i = blockIdx.x * 256 + threadIdx.x;
  if (i < n) out[i] = f2bf(in[i]);
}

// ----------------- fp32 [K][N] -> bf16 transposed [N][K] -------------------
__global__ void __launch_bounds__(256) cvt_transpose_bf16_kernel(const float* __restrict__ in,
                                                                 u16* __restrict__ out,
                                                                 int K, int N) {
  __shared__ u16 t[32][33];
  const int bn = blockIdx.x * 32;
  const int bk = blockIdx.y * 32;
  const int tx = threadIdx.x & 31;
  const int ty = threadIdx.x >> 5;      // 0..7
#pragma unroll
  for (int j = 0; j < 32; j += 8)
    t[ty + j][tx] = f2bf(in[(size_t)(bk + ty + j) * N + bn + tx]);
  __syncthreads();
#pragma unroll
  for (int j = 0; j < 32; j += 8)
    out[(size_t)(bn + ty + j) * K + bk + tx] = t[tx][ty + j];
}

// ------------- bf16 GEMM: C = A * Bt^T + bias (fp32 out) -------------------
// A:  [M,K] row-major bf16.   Bt: [N,K] row-major bf16 (pre-transposed B).
// Block tile 128x128, BK=64, double-buffered LDS with pipelined TDM:
// the DMA for stage i+1 overlaps the 16 WMMAs of stage i.
__global__ void __launch_bounds__(256) gemm_bf16_kernel(const u16* __restrict__ A,
                                                        const u16* __restrict__ Bt,
                                                        const float* __restrict__ bias,
                                                        float* __restrict__ C,
                                                        int M, int N, int K) {
  __shared__ __align__(16) u16 As[2][128][64];
  __shared__ __align__(16) u16 Bs[2][128][64];

  const int tid  = threadIdx.x;
  const int lane = tid & 31;
  const int wid  = tid >> 5;
  const int l16  = lane & 15;
  const int hi   = lane >> 4;
  const int base0 = hi ? 8 : 0;               // A-frag K sub-offset
  const int boff  = hi ? 16 : 0;              // unused (B-frag loads use base0)
  const int row0 = blockIdx.y * 128;
  const int col0 = blockIdx.x * 128;
  const int mrow = 32 * (wid & 3);
  const int ncol = 64 * (wid >> 2);
  const int nStage = K / 64;                  // K always a multiple of 64 here

  v8f acc[2][4];
#pragma unroll
  for (int m = 0; m < 2; ++m)
#pragma unroll
    for (int n = 0; n < 4; ++n)
#pragma unroll
      for (int j = 0; j < 8; ++j) acc[m][n][j] = 0.0f;

#if USE_TDM
  // Prologue: kick off stage 0 into buffer 0.
  if (wid == 0) {
    tdm_load_2d(lds_off_u32(&As[0][0][0]), A + (size_t)row0 * K,
                64, 128, 64, (unsigned)(M - row0), (unsigned long long)K);
    tdm_load_2d(lds_off_u32(&Bs[0][0][0]), Bt + (size_t)col0 * K,
                64, 128, 64, 128, (unsigned long long)K);
  }
#endif

  for (int it = 0; it < nStage; ++it) {
    const int cur = it & 1;
#if USE_TDM
    if (wid == 0) {
      if (it + 1 < nStage) {
        const int k0n = (it + 1) * 64;
        tdm_load_2d(lds_off_u32(&As[cur ^ 1][0][0]), A + (size_t)row0 * K + k0n,
                    64, 128, 64, (unsigned)(M - row0), (unsigned long long)K);
        tdm_load_2d(lds_off_u32(&Bs[cur ^ 1][0][0]), Bt + (size_t)col0 * K + k0n,
                    64, 128, 64, 128, (unsigned long long)K);
        // Current stage's 2 transfers retired; next stage's 2 stay in flight.
        __builtin_amdgcn_s_wait_tensorcnt(2);
      } else {
        __builtin_amdgcn_s_wait_tensorcnt(0);
      }
    }
    __syncthreads();
#else
    const int k0 = it * 64;
    // Manual staging: 4 x 16B chunks per thread per tile.
#pragma unroll
    for (int i = 0; i < 4; ++i) {
      const int c  = tid + i * 256;           // 0..1023
      const int ar = c >> 3;                  // 0..127
      const int ac = (c & 7) * 8;             // 0..56
      u32x4 va = {0u, 0u, 0u, 0u};
      if (row0 + ar < M)
        va = *(const u32x4*)(A + (size_t)(row0 + ar) * K + k0 + ac);
      *(u32x4*)(&As[cur][ar][ac]) = va;
      *(u32x4*)(&Bs[cur][ar][ac]) =
          *(const u32x4*)(Bt + (size_t)(col0 + ar) * K + k0 + ac);
    }
    __syncthreads();
#endif

#pragma unroll
    for (int ks = 0; ks < 2; ++ks) {
      FragU af[2];
#pragma unroll
      for (int m = 0; m < 2; ++m) {
        af[m].q[0] = *(const u32x4*)(&As[cur][mrow + 16 * m + l16][32 * ks + base0]);
        af[m].q[1] = *(const u32x4*)(&As[cur][mrow + 16 * m + l16][32 * ks + 16 + base0]);
      }
#pragma unroll
      for (int nt = 0; nt < 4; ++nt) {
        FragU bf;
        bf.q[0] = *(const u32x4*)(&Bs[cur][ncol + 16 * nt + l16][32 * ks + base0]);
        bf.q[1] = *(const u32x4*)(&Bs[cur][ncol + 16 * nt + l16][32 * ks + 16 + base0]);
        acc[0][nt] = __builtin_amdgcn_wmma_f32_16x16x32_bf16(
            false, af[0].v, false, bf.v, (short)0, acc[0][nt], false, false);
        acc[1][nt] = __builtin_amdgcn_wmma_f32_16x16x32_bf16(
            false, af[1].v, false, bf.v, (short)0, acc[1][nt], false, false);
      }
    }
    __syncthreads();
  }

#pragma unroll
  for (int nt = 0; nt < 4; ++nt) {
    const int nc = col0 + ncol + 16 * nt + l16;
    const float bb = bias ? bias[nc] : 0.0f;
#pragma unroll
    for (int m = 0; m < 2; ++m)
#pragma unroll
      for (int r = 0; r < 8; ++r) {
        const int gr = row0 + mrow + 16 * m + r + 8 * hi;
        if (gr < M) C[(size_t)gr * N + nc] = acc[m][nt][r] + bb;
      }
  }
}

// --------------------- RMSNorm (full 1536 dim) + RoPE ----------------------
__global__ void __launch_bounds__(256) normrope_kernel(const float* __restrict__ qkv,
                                                       const float* __restrict__ fcos,
                                                       const float* __restrict__ fsin,
                                                       const float* __restrict__ gq,
                                                       const float* __restrict__ gk,
                                                       u16* __restrict__ Qh,
                                                       u16* __restrict__ Kh,
                                                       u16* __restrict__ Vh) {
  const int s = blockIdx.x;
  const int tid = threadIdx.x;
  const float* row = qkv + (size_t)s * NQKV_;
  __shared__ float red[256];

  float sq = 0.0f, sk = 0.0f;
#pragma unroll
  for (int j = 0; j < 6; ++j) {
    const int i = tid + j * 256;
    const float qv = row[i];          sq += qv * qv;
    const float kv = row[DIM_ + i];   sk += kv * kv;
  }
  red[tid] = sq; __syncthreads();
  for (int st = 128; st > 0; st >>= 1) { if (tid < st) red[tid] += red[tid + st]; __syncthreads(); }
  const float qss = red[0]; __syncthreads();
  red[tid] = sk; __syncthreads();
  for (int st = 128; st > 0; st >>= 1) { if (tid < st) red[tid] += red[tid + st]; __syncthreads(); }
  const float kss = red[0];

  const float rq = rsqrtf(qss * (1.0f / DIM_) + 1e-6f);
  const float rk = rsqrtf(kss * (1.0f / DIM_) + 1e-6f);

#pragma unroll
  for (int j = 0; j < 3; ++j) {
    const int pp = tid + j * 256;
    const int hh = pp >> 6;
    const int pi = pp & 63;
    const int i0 = hh * HD_ + 2 * pi;
    const float c  = fcos[(size_t)s * 64 + pi];
    const float sn = fsin[(size_t)s * 64 + pi];
    const size_t ob = ((size_t)hh * SEQ_ + s) * HD_ + 2 * pi;

    const float x1 = row[i0]     * rq * gq[i0];
    const float x2 = row[i0 + 1] * rq * gq[i0 + 1];
    Qh[ob]     = f2bf(x1 * c - x2 * sn);
    Qh[ob + 1] = f2bf(x1 * sn + x2 * c);

    const float y1 = row[DIM_ + i0]     * rk * gk[i0];
    const float y2 = row[DIM_ + i0 + 1] * rk * gk[i0 + 1];
    Kh[ob]     = f2bf(y1 * c - y2 * sn);
    Kh[ob + 1] = f2bf(y1 * sn + y2 * c);
  }
#pragma unroll
  for (int j = 0; j < 6; ++j) {
    const int i = tid + j * 256;
    const int hh = i >> 7, dd = i & 127;
    Vh[((size_t)hh * SEQ_ + s) * HD_ + dd] = f2bf(row[2 * DIM_ + i]);
  }
}

// --------------------------- flash attention --------------------------------
// grid: (q-blocks of 128 rows, heads). Block = 8 waves; wave owns 16 q rows.
__global__ void __launch_bounds__(256) attn_kernel(const u16* __restrict__ Qh,
                                                   const u16* __restrict__ Kh,
                                                   const u16* __restrict__ Vh,
                                                   u16* __restrict__ O) {
  __shared__ __align__(16) u16 Ks[32][128];      // [key][d]  (TDM target)
  __shared__ __align__(16) u16 Vs[128][32];      // [d][key]  (transposed)
  __shared__ __align__(16) u16 Ps[8][16][32];    // per-wave probs tile

  const int tid  = threadIdx.x;
  const int lane = tid & 31;
  const int wid  = tid >> 5;
  const int l16  = lane & 15;
  const int hi   = lane >> 4;
  const int base0 = hi ? 8 : 0;
  const int boff  = hi ? 16 : 0;
  const int h  = blockIdx.y;
  const int q0 = blockIdx.x * 128;
  const int qrow0 = q0 + wid * 16;
  const float SCALE = 0.08838834764831845f;      // 1/sqrt(128)

  int qs = qrow0 + l16; if (qs > SEQ_ - 1) qs = SEQ_ - 1;
  const u16* qp = Qh + ((size_t)h * SEQ_ + qs) * HD_;
  v16bf qf[4];
#pragma unroll
  for (int d = 0; d < 4; ++d) {
    FragU u;
    u.q[0] = *(const u32x4*)(qp + 32 * d + base0);
    u.q[1] = *(const u32x4*)(qp + 32 * d + 16 + base0);
    qf[d] = u.v;
  }

  v8f acc[8];
#pragma unroll
  for (int i = 0; i < 8; ++i)
#pragma unroll
    for (int j = 0; j < 8; ++j) acc[i][j] = 0.0f;

  float mprev[8], lsum[8];
  int qfr[8];
#pragma unroll
  for (int r = 0; r < 8; ++r) {
    mprev[r] = -3.0e38f; lsum[r] = 0.0f;
    int gq = qrow0 + r + 8 * hi; if (gq > SEQ_ - 1) gq = SEQ_ - 1;
    qfr[r] = gq / FRAME_;
  }

  const int qlast = imin(q0 + 127, SEQ_ - 1);
  const int kmax  = imin(SEQ_, ((qlast / FRAME_) + 1) * FRAME_);
  const int nkt   = (kmax + 31) / 32;

  for (int kt = 0; kt < nkt; ++kt) {
    const int kb = kt * 32;
#if USE_TDM
    if (wid == 0) {
      tdm_load_2d(lds_off_u32(&Ks[0][0]), Kh + ((size_t)h * SEQ_ + kb) * HD_,
                  HD_, 32, HD_, (unsigned)(SEQ_ - kb), (unsigned long long)HD_);
      __builtin_amdgcn_s_wait_tensorcnt(0);
    }
#pragma unroll
    for (int j = 0; j < 16; ++j) {
      const int flat = tid + j * 256;
      const int key = flat >> 7;
      const int dd  = flat & 127;
      const int gk  = kb + key;
      Vs[dd][key] = (gk < SEQ_)
          ? Vh[((size_t)h * SEQ_ + gk) * HD_ + dd] : (u16)0;
    }
#else
#pragma unroll
    for (int j = 0; j < 16; ++j) {
      const int flat = tid + j * 256;
      const int key = flat >> 7;
      const int dd  = flat & 127;
      const int gk  = kb + key;
      const size_t gidx = ((size_t)h * SEQ_ + (gk < SEQ_ ? gk : SEQ_ - 1)) * HD_ + dd;
      Ks[key][dd] = (gk < SEQ_) ? Kh[gidx] : (u16)0;
      Vs[dd][key] = (gk < SEQ_) ? Vh[gidx] : (u16)0;
    }
#endif
    __syncthreads();

    // scores: two 16x16 tiles over 32 keys, contraction over d (4 x K=32).
    v8f s0, s1;
#pragma unroll
    for (int j = 0; j < 8; ++j) { s0[j] = 0.0f; s1[j] = 0.0f; }
#pragma unroll
    for (int d = 0; d < 4; ++d) {
      FragU b0, b1;
      b0.q[0] = *(const u32x4*)(&Ks[l16][32 * d + base0]);
      b0.q[1] = *(const u32x4*)(&Ks[l16][32 * d + 16 + base0]);
      b1.q[0] = *(const u32x4*)(&Ks[16 + l16][32 * d + base0]);
      b1.q[1] = *(const u32x4*)(&Ks[16 + l16][32 * d + 16 + base0]);
      s0 = __builtin_amdgcn_wmma_f32_16x16x32_bf16(false, qf[d], false, b0.v, (short)0, s0, false, false);
      s1 = __builtin_amdgcn_wmma_f32_16x16x32_bf16(false, qf[d], false, b1.v, (short)0, s1, false, false);
    }

    // block-causal mask + online softmax (16-lane xor reductions).
    float alpha[8];
    const int k0i = kb + l16;
    const int k1i = k0i + 16;
#pragma unroll
    for (int r = 0; r < 8; ++r) {
      float v0 = s0[r] * SCALE;
      float v1 = s1[r] * SCALE;
      if (k0i >= SEQ_ || (k0i / FRAME_) > qfr[r]) v0 = -1e30f;
      if (k1i >= SEQ_ || (k1i / FRAME_) > qfr[r]) v1 = -1e30f;
      float mloc = fmaxf(v0, v1);
      mloc = fmaxf(mloc, __shfl_xor(mloc, 1, 32));
      mloc = fmaxf(mloc, __shfl_xor(mloc, 2, 32));
      mloc = fmaxf(mloc, __shfl_xor(mloc, 4, 32));
      mloc = fmaxf(mloc, __shfl_xor(mloc, 8, 32));
      const float mnew = fmaxf(mprev[r], mloc);
      const float a = __expf(mprev[r] - mnew);
      alpha[r] = a; mprev[r] = mnew;
      const float p0 = __expf(v0 - mnew);
      const float p1 = __expf(v1 - mnew);
      float rs = p0 + p1;
      rs += __shfl_xor(rs, 1, 32);
      rs += __shfl_xor(rs, 2, 32);
      rs += __shfl_xor(rs, 4, 32);
      rs += __shfl_xor(rs, 8, 32);
      lsum[r] = lsum[r] * a + rs;
      const int rowl = r + 8 * hi;
      Ps[wid][rowl][l16]      = f2bf(p0);
      Ps[wid][rowl][l16 + 16] = f2bf(p1);
    }
#pragma unroll
    for (int nt = 0; nt < 8; ++nt)
#pragma unroll
      for (int r = 0; r < 8; ++r) acc[nt][r] *= alpha[r];

    FragU pu;
    pu.q[0] = *(const u32x4*)(&Ps[wid][l16][base0]);
    pu.q[1] = *(const u32x4*)(&Ps[wid][l16][16 + base0]);

#pragma unroll
    for (int nt = 0; nt < 8; ++nt) {
      const int col = nt * 16 + l16;
      FragU vu;
      vu.q[0] = *(const u32x4*)(&Vs[col][base0]);
      vu.q[1] = *(const u32x4*)(&Vs[col][16 + base0]);
      acc[nt] = __builtin_amdgcn_wmma_f32_16x16x32_bf16(
          false, pu.v, false, vu.v, (short)0, acc[nt], false, false);
    }
    __syncthreads();
  }

#pragma unroll
  for (int nt = 0; nt < 8; ++nt) {
    const int col = h * HD_ + nt * 16 + l16;
#pragma unroll
    for (int r = 0; r < 8; ++r) {
      const int gr = qrow0 + r + 8 * hi;
      if (gr < SEQ_) O[(size_t)gr * DIM_ + col] = f2bf(acc[nt][r] / lsum[r]);
    }
  }
}

// ---------------------------------------------------------------------------
extern "C" void kernel_launch(void* const* d_in, const int* in_sizes, int n_in,
                              void* d_out, int out_size, void* d_ws, size_t ws_size,
                              hipStream_t stream) {
  (void)in_sizes; (void)n_in; (void)out_size; (void)ws_size;
  const float* hs   = (const float*)d_in[0];
  const float* fcos = (const float*)d_in[1];
  const float* fsin = (const float*)d_in[2];
  const float* wqkv = (const float*)d_in[3];
  const float* bqkv = (const float*)d_in[4];
  const float* gq   = (const float*)d_in[5];
  const float* gk   = (const float*)d_in[6];
  const float* wout = (const float*)d_in[7];
  const float* bout = (const float*)d_in[8];
  float* out = (float*)d_out;

  char* ws = (char*)d_ws;
  size_t off = 0;
  auto carve = [&](size_t bytes) -> void* {
    void* p = ws + off;
    off += (bytes + 255) & ~(size_t)255;
    return p;
  };
  u16*   hs_bf    = (u16*)carve((size_t)SEQ_ * DIM_ * 2);
  u16*   wqkv_bft = (u16*)carve((size_t)DIM_ * NQKV_ * 2);  // [N=4608][K=1536]
  u16*   wout_bft = (u16*)carve((size_t)DIM_ * DIM_ * 2);   // [N=1536][K=1536]
  float* qkv      = (float*)carve((size_t)SEQ_ * NQKV_ * 4);
  u16*   Qh       = (u16*)carve((size_t)HEADS_ * SEQ_ * HD_ * 2);
  u16*   Kh       = (u16*)carve((size_t)HEADS_ * SEQ_ * HD_ * 2);
  u16*   Vh       = (u16*)carve((size_t)HEADS_ * SEQ_ * HD_ * 2);
  u16*   attn     = (u16*)carve((size_t)SEQ_ * DIM_ * 2);

  // 1) conversions: activations plain, weights transposed to [N][K] bf16.
  {
    int n = SEQ_ * DIM_;
    cvt_bf16_kernel<<<(n + 255) / 256, 256, 0, stream>>>(hs, hs_bf, n);
    cvt_transpose_bf16_kernel<<<dim3(NQKV_ / 32, DIM_ / 32), 256, 0, stream>>>(
        wqkv, wqkv_bft, DIM_, NQKV_);
    cvt_transpose_bf16_kernel<<<dim3(DIM_ / 32, DIM_ / 32), 256, 0, stream>>>(
        wout, wout_bft, DIM_, DIM_);
  }
  // 2) QKV projection: [3120,1536] x [1536,4608] + b_qkv -> fp32.
  gemm_bf16_kernel<<<dim3(NQKV_ / 128, (SEQ_ + 127) / 128), 256, 0, stream>>>(
      hs_bf, wqkv_bft, bqkv, qkv, SEQ_, NQKV_, DIM_);
  // 3) RMSNorm + RoPE -> head-major bf16 Q/K/V.
  normrope_kernel<<<SEQ_, 256, 0, stream>>>(qkv, fcos, fsin, gq, gk, Qh, Kh, Vh);
  // 4) block-causal flash attention.
  attn_kernel<<<dim3((SEQ_ + 127) / 128, HEADS_), 256, 0, stream>>>(Qh, Kh, Vh, attn);
  // 5) output projection: [3120,1536] x [1536,1536] + b_out -> fp32 d_out.
  gemm_bf16_kernel<<<dim3(DIM_ / 128, (SEQ_ + 127) / 128), 256, 0, stream>>>(
      attn, wout_bft, bout, out, SEQ_, DIM_, DIM_);
}
